// SlotAttention_35905926595054
// MI455X (gfx1250) — compile-verified, matrix operand order checked by reading, not run on hardware
//
#include <hip/hip_runtime.h>
#include <cstddef>
#include <cstdint>

// ---------------------------------------------------------------------------
// CDNA5 (gfx1250, wave32) WMMA types: V_WMMA_F32_16X16X32_BF16
// ---------------------------------------------------------------------------
typedef __attribute__((ext_vector_type(16))) __bf16 v16bf;
typedef __attribute__((ext_vector_type(8)))  __bf16 v8bf;
typedef __attribute__((ext_vector_type(8)))  float  v8f;

#define ATT_SCALE 0.03608439182435161f   // 768^-0.5
#define LN_EPS    1e-5f

static constexpr int kS = 4096;    // slots
static constexpr int kN = 16384;   // inputs
static constexpr int kD = 768;     // feature dim
static constexpr int kH = 768;     // mlp hidden
static constexpr int kNCH = 8;     // N-chunks for pass A parallelism

// Async global->LDS (ASYNCcnt-tracked DMA) if the toolchain exposes it.
#if defined(__HIP_DEVICE_COMPILE__) &&                                         \
    __has_builtin(__builtin_amdgcn_global_load_async_to_lds_b128) &&           \
    __has_builtin(__builtin_amdgcn_s_wait_asynccnt)
#define USE_ASYNC_LDS 1
typedef int v4i_vs __attribute__((vector_size(16)));
typedef __attribute__((address_space(1))) v4i_vs* g_v4i_p;   // global int4*
typedef __attribute__((address_space(3))) v4i_vs* l_v4i_p;   // LDS int4*
#else
#define USE_ASYNC_LDS 0
#endif

static __device__ __forceinline__ v8f vzero8() {
  v8f z;
#pragma unroll
  for (int i = 0; i < 8; ++i) z[i] = 0.0f;
  return z;
}

static __device__ __forceinline__ unsigned short bfbits(float f) {
  union { float f; unsigned u; } a; a.f = f;
  unsigned u = a.u + 0x7FFFu + ((a.u >> 16) & 1u);   // round-to-nearest-even
  return (unsigned short)(u >> 16);
}
static __device__ __forceinline__ __bf16 f2bf(float f) {
  unsigned short h = bfbits(f);
  __bf16 r; __builtin_memcpy(&r, &h, 2);
  return r;
}

static __device__ __forceinline__ v8f wmma_bf16(v16bf a, v16bf b, v8f c) {
  return __builtin_amdgcn_wmma_f32_16x16x32_bf16(false, a, false, b, (short)0, c,
                                                 false, false);
}

// A/B fragment from a row-major source (works for global or LDS pointers).
// lane l -> row (l&15); kb=(l>>4)*8; elems[0..7]=cols c0+kb.., [8..15]=c0+16+kb..
static __device__ __forceinline__ v16bf load_frag_rm(const __bf16* base, int ld,
                                                     int r0, int c0) {
  const int l = threadIdx.x & 31;
  const __bf16* p = base + (size_t)(r0 + (l & 15)) * ld + c0 + ((l >> 4) << 3);
  v8bf lo = *(const v8bf*)(p);
  v8bf hi = *(const v8bf*)(p + 16);
  v16bf f;
#pragma unroll
  for (int i = 0; i < 8; ++i) { f[i] = lo[i]; f[8 + i] = hi[i]; }
  return f;
}

// ---------------------------------------------------------------------------
// Elementwise helpers
// ---------------------------------------------------------------------------
__global__ __launch_bounds__(256) void cvt_f32_bf16(const float* __restrict__ src,
                                                    __bf16* __restrict__ dst,
                                                    size_t n) {
  size_t i = (size_t)blockIdx.x * 256 + threadIdx.x;
  size_t stride = (size_t)gridDim.x * 256;
  for (; i < n; i += stride) dst[i] = f2bf(src[i]);
}

__global__ __launch_bounds__(256) void ln_bf16(const float* __restrict__ x,
                                               const float* __restrict__ g,
                                               const float* __restrict__ b,
                                               __bf16* __restrict__ out) {
  const int row = blockIdx.x, t = threadIdx.x;
  const float* xr = x + (size_t)row * kD;
  float s = 0.f, s2 = 0.f;
  for (int i = t; i < kD; i += 256) { float v = xr[i]; s += v; s2 += v * v; }
  __shared__ float rs[256], rs2[256];
  rs[t] = s; rs2[t] = s2; __syncthreads();
  for (int o = 128; o > 0; o >>= 1) {
    if (t < o) { rs[t] += rs[t + o]; rs2[t] += rs2[t + o]; }
    __syncthreads();
  }
  const float mean = rs[0] / kD;
  const float var  = rs2[0] / kD - mean * mean;
  const float inv  = rsqrtf(var + LN_EPS);
  __bf16* orow = out + (size_t)row * kD;
  for (int i = t; i < kD; i += 256)
    orow[i] = f2bf((xr[i] - mean) * inv * g[i] + b[i]);
}

__global__ __launch_bounds__(256) void gru_gate(const float* __restrict__ gx,
                                                const float* __restrict__ gh,
                                                const float* __restrict__ sprev,
                                                float* __restrict__ h) {
  size_t i = (size_t)blockIdx.x * 256 + threadIdx.x;
  if (i >= (size_t)kS * kD) return;
  size_t row = i / kD; int col = (int)(i % kD);
  const float* gxr = gx + row * (size_t)(3 * kD);
  const float* ghr = gh + row * (size_t)(3 * kD);
  float xr = gxr[col], xz = gxr[kD + col], xn = gxr[2 * kD + col];
  float hr = ghr[col], hz = ghr[kD + col], hn = ghr[2 * kD + col];
  float r = 1.f / (1.f + __expf(-(xr + hr)));
  float z = 1.f / (1.f + __expf(-(xz + hz)));
  float n = tanhf(xn + r * hn);
  h[i] = (1.f - z) * n + z * sprev[i];
}

// Merge per-chunk softmax stats: m = max_c m_c ; l = sum_c l_c * exp(m_c - m)
__global__ __launch_bounds__(256) void softmax_combine(const float* __restrict__ mp,
                                                       const float* __restrict__ lp,
                                                       float* __restrict__ mg,
                                                       float* __restrict__ lg) {
  int i = blockIdx.x * 256 + threadIdx.x;
  if (i >= kS) return;
  float m = -INFINITY;
#pragma unroll
  for (int c = 0; c < kNCH; ++c) m = fmaxf(m, mp[(size_t)c * kS + i]);
  float l = 0.f;
#pragma unroll
  for (int c = 0; c < kNCH; ++c)
    l += lp[(size_t)c * kS + i] * __expf(mp[(size_t)c * kS + i] - m);
  mg[i] = m; lg[i] = l;
}

// ---------------------------------------------------------------------------
// Generic WMMA GEMM: C[M,NO] = epi(A[M,K] @ W[NO,K]^T + bias [+ resid])
// One wave per 16x64 output tile (one A fragment feeds 4 WMMAs).
// WBT: write bf16 transposed (outB[col*ldT + row], packed b128 stores).
// ---------------------------------------------------------------------------
template <bool RELU, bool RESID, bool WF32, bool WBF, bool WBT>
__global__ __launch_bounds__(32) void gemm_bt(const __bf16* __restrict__ A,
                                              const __bf16* __restrict__ W,
                                              const float* __restrict__ bias,
                                              const float* __restrict__ resid,
                                              float* __restrict__ outF,
                                              __bf16* __restrict__ outB,
                                              int K, int NO, int ldT) {
  const int r0 = blockIdx.x << 4;
  const int c0 = blockIdx.y << 6;
  const int l = threadIdx.x & 31;
  v8f acc[4] = {vzero8(), vzero8(), vzero8(), vzero8()};
  const int nk = K >> 5;
  for (int kk = 0; kk < nk; ++kk) {
    v16bf a = load_frag_rm(A, K, r0, kk << 5);
#pragma unroll
    for (int t = 0; t < 4; ++t)
      acc[t] = wmma_bf16(a, load_frag_rm(W, K, c0 + (t << 4), kk << 5), acc[t]);
  }
  const int rb = r0 + ((l >> 4) << 3);
#pragma unroll
  for (int t = 0; t < 4; ++t) {
    const int col = c0 + (t << 4) + (l & 15);
    const float bv = bias[col];
    float v[8];
#pragma unroll
    for (int r = 0; r < 8; ++r) {
      float x = acc[t][r] + bv;
      if (RELU)  x = fmaxf(x, 0.0f);
      if (RESID) x += resid[(size_t)(rb + r) * NO + col];
      v[r] = x;
    }
    if (WF32) {
#pragma unroll
      for (int r = 0; r < 8; ++r) outF[(size_t)(rb + r) * NO + col] = v[r];
    }
    if (WBF) {
#pragma unroll
      for (int r = 0; r < 8; ++r) outB[(size_t)(rb + r) * NO + col] = f2bf(v[r]);
    }
    if (WBT) {  // transposed store: 8 consecutive rows of one column -> b128
      uint4 u;
      u.x = bfbits(v[0]) | ((unsigned)bfbits(v[1]) << 16);
      u.y = bfbits(v[2]) | ((unsigned)bfbits(v[3]) << 16);
      u.z = bfbits(v[4]) | ((unsigned)bfbits(v[5]) << 16);
      u.w = bfbits(v[6]) | ((unsigned)bfbits(v[7]) << 16);
      *reinterpret_cast<uint4*>(outB + (size_t)col * ldT + rb) = u;
    }
  }
}

// Stage K[n0..n0+31, 0..768) (48KB, contiguous) into LDS with all 256 threads.
// Uses ASYNCcnt-tracked global->LDS DMA when available (bypasses VGPRs),
// otherwise a synchronous b128 copy.
static __device__ __forceinline__ void stage_k(const __bf16* __restrict__ Kb,
                                               __bf16* Klds, int n0) {
  const __bf16* src = Kb + (size_t)n0 * kD;
#if USE_ASYNC_LDS
  __bf16* s = const_cast<__bf16*>(src);
#pragma unroll
  for (int i = 0; i < 12; ++i) {
    const int e = (threadIdx.x + (i << 8)) << 3;  // element idx, 8 bf16 = 16B
    __builtin_amdgcn_global_load_async_to_lds_b128(
        (g_v4i_p)(s + e), (l_v4i_p)(Klds + e), 0, 0);
  }
  __builtin_amdgcn_s_wait_asynccnt(0);
#else
  const uint4* s4 = reinterpret_cast<const uint4*>(src);
  uint4* d4 = reinterpret_cast<uint4*>(Klds);
#pragma unroll
  for (int i = 0; i < 12; ++i)
    d4[threadIdx.x + (i << 8)] = s4[threadIdx.x + (i << 8)];
#endif
}

// ---------------------------------------------------------------------------
// Attention pass A: 8 waves x 16 Q rows share an LDS-staged K tile; each wave
// keeps 24 resident Q fragments and maintains online (m, l) stats over one
// N-chunk. Grid: (S/128, kNCH).
// ---------------------------------------------------------------------------
__global__ __launch_bounds__(256) void attn_stats(const __bf16* __restrict__ Qb,
                                                  const __bf16* __restrict__ Kb,
                                                  float* __restrict__ mp,
                                                  float* __restrict__ lp) {
  __shared__ alignas(16) __bf16 Klds[32 * kD];
  const int w = threadIdx.x >> 5, l = threadIdx.x & 31;
  const int r0 = blockIdx.x * 128 + w * 16;
  const int nbase = blockIdx.y * (kN / kNCH);
  const int nend  = nbase + (kN / kNCH);

  v16bf qf[24];
#pragma unroll
  for (int kk = 0; kk < 24; ++kk) qf[kk] = load_frag_rm(Qb, kD, r0, kk << 5);

  float mrun[8], lrun[8];
#pragma unroll
  for (int r = 0; r < 8; ++r) { mrun[r] = -INFINITY; lrun[r] = 0.0f; }

  for (int n0 = nbase; n0 < nend; n0 += 32) {
    __builtin_prefetch(Kb + (size_t)(n0 + 32) * kD, 0, 1);
    stage_k(Kb, Klds, n0);
    __syncthreads();
    v8f a0 = vzero8(), a1 = vzero8();
#pragma unroll
    for (int kk = 0; kk < 24; ++kk) {
      a0 = wmma_bf16(qf[kk], load_frag_rm(Klds, kD, 0,  kk << 5), a0);
      a1 = wmma_bf16(qf[kk], load_frag_rm(Klds, kD, 16, kk << 5), a1);
    }
    __syncthreads();
#pragma unroll
    for (int r = 0; r < 8; ++r) {
      float d0 = a0[r] * ATT_SCALE, d1 = a1[r] * ATT_SCALE;
      float tm = fmaxf(d0, d1);
      tm = fmaxf(tm, __shfl_xor(tm, 1, 32));
      tm = fmaxf(tm, __shfl_xor(tm, 2, 32));
      tm = fmaxf(tm, __shfl_xor(tm, 4, 32));
      tm = fmaxf(tm, __shfl_xor(tm, 8, 32));
      float mnew = fmaxf(mrun[r], tm);
      float ts = __expf(d0 - mnew) + __expf(d1 - mnew);
      ts += __shfl_xor(ts, 1, 32);
      ts += __shfl_xor(ts, 2, 32);
      ts += __shfl_xor(ts, 4, 32);
      ts += __shfl_xor(ts, 8, 32);
      lrun[r] = lrun[r] * __expf(mrun[r] - mnew) + ts;
      mrun[r] = mnew;
    }
  }
  if ((l & 15) == 0) {
    const int rb = r0 + ((l >> 4) << 3);
#pragma unroll
    for (int r = 0; r < 8; ++r) {
      mp[(size_t)blockIdx.y * kS + rb + r] = mrun[r];
      lp[(size_t)blockIdx.y * kS + rb + r] = lrun[r];
    }
  }
}

// ---------------------------------------------------------------------------
// Attention pass B: recompute logits from the shared LDS K tile, exponentiate
// (final m from pass A), transpose P through a per-wave LDS slot into A-frag
// layout, and accumulate updates[16 x 64] = P @ V using contiguous fragments
// from Vt (= V^T, written transposed by the V projection GEMM).
// Grid: (S/128, D/64), 256 threads.
// ---------------------------------------------------------------------------
__global__ __launch_bounds__(256) void attn_apply(const __bf16* __restrict__ Qb,
                                                  const __bf16* __restrict__ Kb,
                                                  const __bf16* __restrict__ Vt,
                                                  const float* __restrict__ mg,
                                                  const float* __restrict__ lg,
                                                  float* __restrict__ updF,
                                                  __bf16* __restrict__ updB) {
  __shared__ alignas(16) __bf16 Klds[32 * kD];
  __shared__ alignas(16) __bf16 P[8][16 * 32];
  const int w = threadIdx.x >> 5, l = threadIdx.x & 31;
  const int r0 = blockIdx.x * 128 + w * 16;
  const int c0 = blockIdx.y << 6;

  v16bf qf[24];
#pragma unroll
  for (int kk = 0; kk < 24; ++kk) qf[kk] = load_frag_rm(Qb, kD, r0, kk << 5);

  const int rb = (l >> 4) << 3;
  float mr[8];
#pragma unroll
  for (int r = 0; r < 8; ++r) mr[r] = mg[r0 + rb + r];

  v8f acc0 = vzero8(), acc1 = vzero8(), acc2 = vzero8(), acc3 = vzero8();

  for (int n0 = 0; n0 < kN; n0 += 32) {
    __builtin_prefetch(Kb + (size_t)(n0 + 32) * kD, 0, 1);
    stage_k(Kb, Klds, n0);
    __syncthreads();
    v8f d0 = vzero8(), d1 = vzero8();
#pragma unroll
    for (int kk = 0; kk < 24; ++kk) {
      d0 = wmma_bf16(qf[kk], load_frag_rm(Klds, kD, 0,  kk << 5), d0);
      d1 = wmma_bf16(qf[kk], load_frag_rm(Klds, kD, 16, kk << 5), d1);
    }
    const int col = l & 15;
#pragma unroll
    for (int r = 0; r < 8; ++r) {
      P[w][(rb + r) * 32 + col]      = f2bf(__expf(d0[r] * ATT_SCALE - mr[r]));
      P[w][(rb + r) * 32 + 16 + col] = f2bf(__expf(d1[r] * ATT_SCALE - mr[r]));
    }
    // same-wave LDS RAW: DS pipeline is in-order within a wave
    v16bf pf = load_frag_rm(P[w], 32, 0, 0);
    acc0 = wmma_bf16(pf, load_frag_rm(Vt, kN, c0,      n0), acc0);
    acc1 = wmma_bf16(pf, load_frag_rm(Vt, kN, c0 + 16, n0), acc1);
    acc2 = wmma_bf16(pf, load_frag_rm(Vt, kN, c0 + 32, n0), acc2);
    acc3 = wmma_bf16(pf, load_frag_rm(Vt, kN, c0 + 48, n0), acc3);
    __syncthreads();
  }

  float li[8];
#pragma unroll
  for (int r = 0; r < 8; ++r) li[r] = 1.0f / lg[r0 + rb + r];
  const int colb = l & 15;
#pragma unroll
  for (int r = 0; r < 8; ++r) {
    const size_t row = (size_t)(r0 + rb + r);
    float v0 = acc0[r] * li[r], v1 = acc1[r] * li[r];
    float v2 = acc2[r] * li[r], v3 = acc3[r] * li[r];
    updF[row * kD + c0 + colb]      = v0; updB[row * kD + c0 + colb]      = f2bf(v0);
    updF[row * kD + c0 + 16 + colb] = v1; updB[row * kD + c0 + 16 + colb] = f2bf(v1);
    updF[row * kD + c0 + 32 + colb] = v2; updB[row * kD + c0 + 32 + colb] = f2bf(v2);
    updF[row * kD + c0 + 48 + colb] = v3; updB[row * kD + c0 + 48 + colb] = f2bf(v3);
  }
}

// ---------------------------------------------------------------------------
// Host orchestration
// ---------------------------------------------------------------------------
extern "C" void kernel_launch(void* const* d_in, const int* in_sizes, int n_in,
                              void* d_out, int out_size, void* d_ws, size_t ws_size,
                              hipStream_t stream) {
  (void)in_sizes; (void)n_in; (void)out_size; (void)ws_size;
  const int S = kS, N = kN, D = kD, H = kH, ITERS = 3;

  const float* cand  = (const float*)d_in[0];
  const float* pano  = (const float*)d_in[1];
  const float* Wq    = (const float*)d_in[2];
  const float* bq    = (const float*)d_in[3];
  const float* Wk    = (const float*)d_in[4];
  const float* bk    = (const float*)d_in[5];
  const float* Wv    = (const float*)d_in[6];
  const float* bv    = (const float*)d_in[7];
  const float* W_ih  = (const float*)d_in[8];
  const float* b_ih  = (const float*)d_in[9];
  const float* W_hh  = (const float*)d_in[10];
  const float* b_hh  = (const float*)d_in[11];
  const float* W1    = (const float*)d_in[12];
  const float* b1    = (const float*)d_in[13];
  const float* W2    = (const float*)d_in[14];
  const float* b2    = (const float*)d_in[15];
  const float* g_in  = (const float*)d_in[16];
  const float* be_in = (const float*)d_in[17];
  const float* g_sl  = (const float*)d_in[18];
  const float* be_sl = (const float*)d_in[19];
  const float* g_ff  = (const float*)d_in[20];
  const float* be_ff = (const float*)d_in[21];

  float* slots = (float*)d_out;  // fp32 slots live in d_out across iterations

  char* ws = (char*)d_ws;
  size_t off = 0;
  auto carve = [&](size_t bytes) -> void* {
    void* p = ws + off;
    off = (off + bytes + 255) & ~(size_t)255;
    return p;
  };
  __bf16* panoln = (__bf16*)carve((size_t)N * D * 2);
  __bf16* Kb     = (__bf16*)carve((size_t)N * D * 2);
  __bf16* Vt     = (__bf16*)carve((size_t)D * N * 2);   // V^T  [D, N]
  __bf16* slotln = (__bf16*)carve((size_t)S * D * 2);
  __bf16* Qb     = (__bf16*)carve((size_t)S * D * 2);
  __bf16* sprevb = (__bf16*)carve((size_t)S * D * 2);
  __bf16* updb   = (__bf16*)carve((size_t)S * D * 2);
  __bf16* ffb    = (__bf16*)carve((size_t)S * D * 2);
  __bf16* tb     = (__bf16*)carve((size_t)S * H * 2);
  __bf16* Wqb    = (__bf16*)carve((size_t)D * D * 2);
  __bf16* Wkb    = (__bf16*)carve((size_t)D * D * 2);
  __bf16* Wvb    = (__bf16*)carve((size_t)D * D * 2);
  __bf16* Wihb   = (__bf16*)carve((size_t)3 * D * D * 2);
  __bf16* Whhb   = (__bf16*)carve((size_t)3 * D * D * 2);
  __bf16* W1b    = (__bf16*)carve((size_t)H * D * 2);
  __bf16* W2b    = (__bf16*)carve((size_t)D * H * 2);
  float*  mp     = (float*)carve((size_t)kNCH * S * 4);
  float*  lp     = (float*)carve((size_t)kNCH * S * 4);
  float*  mgl    = (float*)carve((size_t)S * 4);
  float*  lgl    = (float*)carve((size_t)S * 4);
  float*  updF   = (float*)carve((size_t)S * D * 4);
  float*  gx     = (float*)carve((size_t)S * 3 * D * 4);
  float*  gh     = (float*)carve((size_t)S * 3 * D * 4);
  float*  hbuf   = (float*)carve((size_t)S * D * 4);

  // 1) weights -> bf16
  cvt_f32_bf16<<<1024, 256, 0, stream>>>(Wq,   Wqb,  (size_t)D * D);
  cvt_f32_bf16<<<1024, 256, 0, stream>>>(Wk,   Wkb,  (size_t)D * D);
  cvt_f32_bf16<<<1024, 256, 0, stream>>>(Wv,   Wvb,  (size_t)D * D);
  cvt_f32_bf16<<<2048, 256, 0, stream>>>(W_ih, Wihb, (size_t)3 * D * D);
  cvt_f32_bf16<<<2048, 256, 0, stream>>>(W_hh, Whhb, (size_t)3 * D * D);
  cvt_f32_bf16<<<1024, 256, 0, stream>>>(W1,   W1b,  (size_t)H * D);
  cvt_f32_bf16<<<1024, 256, 0, stream>>>(W2,   W2b,  (size_t)D * H);

  // 2) slots := cand_feat
  (void)hipMemcpyAsync(slots, cand, (size_t)S * D * 4, hipMemcpyDeviceToDevice,
                       stream);

  // 3) pano LN; K projection (row-major bf16); V projection written transposed
  ln_bf16<<<N, 256, 0, stream>>>(pano, g_in, be_in, panoln);
  gemm_bt<false, false, false, true, false><<<dim3(N / 16, D / 64), 32, 0, stream>>>(
      panoln, Wkb, bk, nullptr, nullptr, Kb, D, D, 0);
  gemm_bt<false, false, false, false, true><<<dim3(N / 16, D / 64), 32, 0, stream>>>(
      panoln, Wvb, bv, nullptr, nullptr, Vt, D, D, N);

  for (int it = 0; it < ITERS; ++it) {
    // q = LN(slots) @ Wq^T + bq
    ln_bf16<<<S, 256, 0, stream>>>(slots, g_sl, be_sl, slotln);
    gemm_bt<false, false, false, true, false><<<dim3(S / 16, D / 64), 32, 0, stream>>>(
        slotln, Wqb, bq, nullptr, nullptr, Qb, D, D, 0);
    cvt_f32_bf16<<<4096, 256, 0, stream>>>(slots, sprevb, (size_t)S * D);

    // two-pass attention; K streamed through 48KB LDS tiles, V^T fragments
    attn_stats<<<dim3(S / 128, kNCH), 256, 0, stream>>>(Qb, Kb, mp, lp);
    softmax_combine<<<(S + 255) / 256, 256, 0, stream>>>(mp, lp, mgl, lgl);
    attn_apply<<<dim3(S / 128, D / 64), 256, 0, stream>>>(Qb, Kb, Vt, mgl, lgl,
                                                          updF, updb);

    // GRU
    gemm_bt<false, false, true, false, false><<<dim3(S / 16, 3 * D / 64), 32, 0, stream>>>(
        updb, Wihb, b_ih, nullptr, gx, nullptr, D, 3 * D, 0);
    gemm_bt<false, false, true, false, false><<<dim3(S / 16, 3 * D / 64), 32, 0, stream>>>(
        sprevb, Whhb, b_hh, nullptr, gh, nullptr, D, 3 * D, 0);
    gru_gate<<<(S * D + 255) / 256, 256, 0, stream>>>(gx, gh, slots, hbuf);

    // MLP with residual: slots = h + relu(LN(h) @ W1^T + b1) @ W2^T + b2
    ln_bf16<<<S, 256, 0, stream>>>(hbuf, g_ff, be_ff, ffb);
    gemm_bt<true, false, false, true, false><<<dim3(S / 16, H / 64), 32, 0, stream>>>(
        ffb, W1b, b1, nullptr, nullptr, tb, D, H, 0);
    gemm_bt<false, true, true, false, false><<<dim3(S / 16, D / 64), 32, 0, stream>>>(
        tb, W2b, b2, hbuf, slots, nullptr, H, D, 0);
  }
}
